// VectorQuantizerMaxEnt_30142080483338
// MI455X (gfx1250) — compile-verified, matrix operand order checked by reading, not run on hardware
//
#include <hip/hip_runtime.h>
#include <hip/hip_bf16.h>

typedef __bf16 bf16;
typedef __attribute__((ext_vector_type(16))) __bf16 v16bf;
typedef __attribute__((ext_vector_type(8)))  __bf16 v8bf;
typedef __attribute__((ext_vector_type(8)))  float  v8f;
typedef __attribute__((ext_vector_type(4)))  int    v4i;

#define N_ROWS 8192
#define N_CODE 8192
#define DIM    512
#define MTILE  32
#define NSAMP  10
#define BSTRIDE 80                    // staged B row stride (bytes): 16B aligned, bank-conflict-free
#define BBUF_BYTES (64 * BSTRIDE)     // 64 codes x 80B = 5120B per buffer

// ---------------------------------------------------------------------------
// CDNA5 16-bit WMMA fragment from a row-major row:
// lanes 0-15: K = {k..k+7} ++ {k+16..k+23}; lanes 16-31: +8. Two b128 loads.
// ---------------------------------------------------------------------------
__device__ __forceinline__ v16bf load_frag(const bf16* rowptr, int k, int half) {
    const v8bf lo = *reinterpret_cast<const v8bf*>(rowptr + k + half * 8);
    const v8bf hi = *reinterpret_cast<const v8bf*>(rowptr + k + 16 + half * 8);
    return __builtin_shufflevector(lo, hi, 0,1,2,3,4,5,6,7,8,9,10,11,12,13,14,15);
}

// Async global->LDS staging of one 64-code x 32-d bf16 chunk (64B per code row).
// Fully coalesced: 4 lanes cover one 64B row segment, 8 rows per instruction.
__device__ __forceinline__ void stage_b(unsigned lds_off, const char* gsrc, int lane) {
    const int cs = lane >> 2;            // code row 0..7
    const int bo = (lane & 3) * 16;      // byte chunk within row
    unsigned dst = lds_off + (unsigned)(cs * BSTRIDE + bo);
    unsigned long long src = (unsigned long long)(gsrc + (size_t)cs * (DIM * 2) + bo);
    #pragma unroll
    for (int i = 0; i < 8; ++i) {        // 8 x 8 rows = 64 codes
        asm volatile("global_load_async_to_lds_b128 %0, %1, off"
                     :: "v"(dst), "v"(src) : "memory");
        dst += 8u * BSTRIDE;
        src += 8ull * (DIM * 2);
    }
}

__device__ __forceinline__ unsigned hash_u32(unsigned x) {
    x ^= x >> 16; x *= 0x7feb352dU;
    x ^= x >> 15; x *= 0x846ca68bU;
    x ^= x >> 16;
    return x;
}

// ---------------------------------------------------------------------------
// Kernel 1: f32 -> bf16 conversion + row squared-norms (one block per row).
// ---------------------------------------------------------------------------
__global__ __launch_bounds__(256) void vq_prep(const float* __restrict__ src,
                                               bf16* __restrict__ dstb,
                                               float* __restrict__ dst2) {
    const int row = blockIdx.x;
    const float* r = src + (size_t)row * DIM;
    bf16* o = dstb + (size_t)row * DIM;
    float ss = 0.f;
    for (int c = threadIdx.x; c < DIM; c += 256) {
        float v = r[c];
        o[c] = (bf16)v;
        ss += v * v;
    }
    #pragma unroll
    for (int m = 16; m >= 1; m >>= 1) ss += __shfl_xor(ss, m, 32);
    __shared__ float ws[8];
    if ((threadIdx.x & 31) == 0) ws[threadIdx.x >> 5] = ss;
    __syncthreads();
    if (threadIdx.x == 0) {
        float t = 0.f;
        #pragma unroll
        for (int i = 0; i < 8; ++i) t += ws[i];
        dst2[row] = t;
    }
}

// ---------------------------------------------------------------------------
// Kernel 2: fused WMMA GEMM (async-staged B) + streaming sum-exp +
// inverse-CDF sampling + gather-mean. 8 waves, 32 rows x all 8192 codes.
// ---------------------------------------------------------------------------
__global__ __launch_bounds__(256) void vq_main(
    const bf16*  __restrict__ xb,  const float* __restrict__ x2,
    const bf16*  __restrict__ eb,  const float* __restrict__ e2,
    const float* __restrict__ embeds,
    float* __restrict__ out_q, float* __restrict__ out_s)
{
    __shared__ __align__(16) bf16 xs[MTILE * DIM];            // 32 KB x tile
    __shared__ __align__(16) char bstage[8 * 2 * BBUF_BYTES]; // 80 KB B staging
    __shared__ float stats[MTILE][N_CODE / 32];               // 32 KB per-32-code sumexp
    __shared__ float x2s[MTILE];
    __shared__ float rowS[MTILE];
    __shared__ int   smp[MTILE][NSAMP];

    const int tid  = threadIdx.x;
    const int wave = tid >> 5;
    const int lane = tid & 31;
    const int half = lane >> 4;
    const int l16  = lane & 15;
    const int rg   = wave >> 2;        // 0..1  -> 16-row group
    const int cg   = wave & 3;         // 0..3  -> 64-code group per 256-code step
    const int rowbase = blockIdx.x * MTILE;
    const int wbase = wave * (2 * BBUF_BYTES);   // this wave's staging region

    // ---- phase 0: stage x tile into LDS ----
    {
        const v4i* src = reinterpret_cast<const v4i*>(xb + (size_t)rowbase * DIM);
        v4i* dst = reinterpret_cast<v4i*>(xs);
        for (int i = tid; i < MTILE * DIM / 8; i += 256) dst[i] = src[i];
        if (tid < MTILE) x2s[tid] = x2[rowbase + tid];
    }
    __syncthreads();

    float xr[8];
    #pragma unroll
    for (int v = 0; v < 8; ++v) xr[v] = x2s[rg * 16 + v + 8 * half];

    const bf16* a_base = xs + (size_t)(rg * 16 + l16) * DIM;
    const unsigned wbase_off = (unsigned)(unsigned long long)&bstage[wbase];
    const char* ebb = (const char*)eb;

    // ---- phase 1: WMMA GEMM over all codes ----
    for (int t = 0; t < N_CODE / 256; ++t) {
        const int code0 = t * 256 + cg * 64;
        const char* gbase = ebb + (size_t)code0 * (DIM * 2);

        if (t + 1 < N_CODE / 256)
            __builtin_prefetch(ebb + (size_t)(code0 + 256) * (DIM * 2), 0, 1);

        v8f acc[4];
        #pragma unroll
        for (int j = 0; j < 4; ++j)
            #pragma unroll
            for (int v = 0; v < 8; ++v) acc[j][v] = 0.f;

        stage_b(wbase_off, gbase, lane);                     // k-step 0 -> buffer 0
        #pragma unroll 2
        for (int kk = 0; kk < DIM / 32; ++kk) {              // 16 k-steps
            const int sel = (kk & 1) * BBUF_BYTES;           // constant per unrolled copy
            if (kk + 1 < DIM / 32) {
                stage_b(wbase_off + (unsigned)(BBUF_BYTES - sel),
                        gbase + (kk + 1) * 64, lane);
                asm volatile("s_wait_asynccnt 0x8" ::: "memory");  // buf[kk] ready
            } else {
                asm volatile("s_wait_asynccnt 0x0" ::: "memory");
            }

            // Load A + all four B fragments; first WMMA consumes the LAST
            // loaded fragment so all ds_loads must be issued (and stay live)
            // before the first s_wait_dscnt -> loads clause together, WMMAs
            // then run back-to-back on the XDL pipe.
            v16bf a = load_frag(a_base, kk * 32, half);      // ds_load_b128 x2
            v16bf bfr[4];
            #pragma unroll
            for (int j = 0; j < 4; ++j) {
                const char* rowb = &bstage[wbase + sel + (j * 16 + l16) * BSTRIDE];
                const v8bf lo = *reinterpret_cast<const v8bf*>(rowb + half * 16);
                const v8bf hi = *reinterpret_cast<const v8bf*>(rowb + 32 + half * 16);
                bfr[j] = __builtin_shufflevector(lo, hi,
                            0,1,2,3,4,5,6,7,8,9,10,11,12,13,14,15);
            }
            #pragma unroll
            for (int j = 3; j >= 0; --j) {                   // reverse consume order
                acc[j] = __builtin_amdgcn_wmma_f32_16x16x32_bf16(
                    false, a, false, bfr[j], (short)0, acc[j], false, false);
            }
        }

        // logits -> exp -> reduce pairs of 16x16 subtiles into 32-code stats
        #pragma unroll
        for (int jb = 0; jb < 4; jb += 2) {
            const float e2a = e2[code0 + jb * 16 + l16];
            const float e2b = e2[code0 + (jb + 1) * 16 + l16];
            #pragma unroll
            for (int v = 0; v < 8; ++v) {
                float ra = __expf(-(xr[v] + e2a - 2.0f * acc[jb][v])     * (1.0f / 8192.0f));
                float rb = __expf(-(xr[v] + e2b - 2.0f * acc[jb + 1][v]) * (1.0f / 8192.0f));
                float r = ra + rb;
                r += __shfl_xor(r, 1, 16);
                r += __shfl_xor(r, 2, 16);
                r += __shfl_xor(r, 4, 16);
                r += __shfl_xor(r, 8, 16);
                if (l16 == v)
                    stats[rg * 16 + v + 8 * half][t * 8 + cg * 2 + (jb >> 1)] = r;
            }
        }
    }
    __syncthreads();

    // ---- phase 2: inverse-CDF sampling ----
    if (tid < MTILE) {
        float s = 0.f;
        for (int i = 0; i < N_CODE / 32; ++i) s += stats[tid][i];
        rowS[tid] = s;
    }
    __syncthreads();

    for (int task = tid; task < MTILE * NSAMP; task += 256) {
        const int r = task / NSAMP, s = task % NSAMP;
        const int grow = rowbase + r;
        const unsigned h = hash_u32((unsigned)(grow * 16 + s) + 0x9E3779B9U);
        const float u = ((float)(h >> 8) + 0.5f) * (1.0f / 16777216.0f);
        const float T = u * rowS[r];

        float cum = 0.f;
        int t32 = 0;
        for (; t32 < N_CODE / 32 - 1; ++t32) {
            float nv = cum + stats[r][t32];
            if (nv >= T) break;
            cum = nv;
        }
        const float R = T - cum;

        // recompute the 32 logits of the landed tile (bf16 GEMV, LDS x / L2 e)
        const bf16* xrow = xs + (size_t)r * DIM;
        int pick = t32 * 32 + 31;
        float c2 = 0.f;
        for (int n = 0; n < 32; ++n) {
            const int code = t32 * 32 + n;
            const bf16* erow = eb + (size_t)code * DIM;
            float dot = 0.f;
            #pragma unroll 2
            for (int d = 0; d < DIM; d += 8) {
                v8bf xa = *reinterpret_cast<const v8bf*>(xrow + d);
                v8bf ea = *reinterpret_cast<const v8bf*>(erow + d);
                #pragma unroll
                for (int q = 0; q < 8; ++q) dot += (float)xa[q] * (float)ea[q];
            }
            c2 += __expf(-(x2s[r] + e2[code] - 2.0f * dot) * (1.0f / 8192.0f));
            if (c2 >= R) { pick = code; break; }
        }
        smp[r][s] = pick;
        out_s[(size_t)grow * NSAMP + s] = (float)pick;
    }
    __syncthreads();

    // ---- phase 3: gather-mean of sampled f32 embedding rows ----
    for (int r = 0; r < MTILE; ++r) {
        for (int d = tid; d < DIM; d += 256) {
            float acc = 0.f;
            #pragma unroll
            for (int s = 0; s < NSAMP; ++s)
                acc += embeds[(size_t)smp[r][s] * DIM + d];
            out_q[((size_t)(rowbase + r)) * DIM + d] = acc * (1.0f / NSAMP);
        }
    }
}

extern "C" void kernel_launch(void* const* d_in, const int* in_sizes, int n_in,
                              void* d_out, int out_size, void* d_ws, size_t ws_size,
                              hipStream_t stream) {
    const float* inputs = (const float*)d_in[0];   // (4,2048,512) f32 -> 8192x512
    const float* embeds = (const float*)d_in[1];   // (8192,512) f32

    char* ws = (char*)d_ws;
    bf16*  xb = (bf16*)(ws);                                   // 8 MB
    bf16*  eb = (bf16*)(ws + ((size_t)8  << 20));              // 8 MB
    float* x2 = (float*)(ws + ((size_t)16 << 20));             // 32 KB
    float* e2 = (float*)(ws + ((size_t)16 << 20) + 32768);     // 32 KB

    vq_prep<<<N_ROWS, 256, 0, stream>>>(inputs, xb, x2);
    vq_prep<<<N_CODE, 256, 0, stream>>>(embeds, eb, e2);

    float* out_q = (float*)d_out;                       // 8192*512 quantized
    float* out_s = out_q + (size_t)N_ROWS * DIM;        // 8192*10 samples (as float)

    vq_main<<<N_ROWS / MTILE, 256, 0, stream>>>(xb, x2, eb, e2, embeds, out_q, out_s);
}
